// GATRecommender_6485400617568
// MI455X (gfx1250) — matrix operand order repaired and checked by reading, not compiled, not run on hardware
//
#include <hip/hip_runtime.h>
#include <hip/hip_bf16.h>

typedef __attribute__((ext_vector_type(16))) __bf16 v16bf;
typedef __attribute__((ext_vector_type(8)))  __bf16 v8bf;
typedef __attribute__((ext_vector_type(8)))  float  v8f;
typedef __attribute__((ext_vector_type(4)))  int    v4i;

#define GN 50000
#define GE 800000

// B-panel chunking for LDS staging
#define KP 128
#define SP (KP + 8)   // padded row stride: 272B = 68 dwords, 68 % 64 = 4 -> no bank alias

#if defined(__HIP_DEVICE_COMPILE__) && __has_builtin(__builtin_amdgcn_global_load_async_to_lds_b128)
#define USE_ASYNC_LDS 1
#else
#define USE_ASYNC_LDS 0
#endif

// ---------------------------------------------------------------------------
// fp32 -> bf16 conversion
// ---------------------------------------------------------------------------
__global__ void f32_to_bf16_k(const float* __restrict__ src, __bf16* __restrict__ dst,
                              long long n) {
    long long i = (long long)blockIdx.x * blockDim.x + threadIdx.x;
    if (i < n) dst[i] = (__bf16)src[i];
}

// ---------------------------------------------------------------------------
// fill u32
// ---------------------------------------------------------------------------
__global__ void fill_u32_k(unsigned* __restrict__ p, unsigned v, long long n) {
    long long i = (long long)blockIdx.x * blockDim.x + threadIdx.x;
    if (i < n) p[i] = v;
}

// ---------------------------------------------------------------------------
// WMMA GEMM: A[M x K](bf16) @ W[Nc x K](bf16)^T + bias -> C fp32 [M x Nc]
// block = 256 threads = 8 waves; block tile = 128 rows x 64 cols
// B panel staged through LDS in 128-wide K chunks (shared by all 8 waves);
// A fragments software-pipelined (next k-step loads issued before current MACs)
// ---------------------------------------------------------------------------
__global__ __launch_bounds__(256)
void gemm_bf16_wmma_k(const __bf16* __restrict__ A, const __bf16* __restrict__ W,
                      const float* __restrict__ bias, float* __restrict__ C,
                      int M, int K, int Nc) {
    __shared__ __bf16 sB[64 * SP];

    const int wave = threadIdx.x >> 5;
    const int lane = threadIdx.x & 31;
    const int half = lane >> 4;       // K-half selector for A / B fragments
    const int r    = lane & 15;

    const int row0 = blockIdx.x * 128 + wave * 16;
    const int col0 = blockIdx.y * 64;

    int arow = row0 + r;
    if (arow >= M) arow = M - 1;      // clamp reads, mask stores (EXEC stays full)
    const __bf16* __restrict__ Arow = A + (size_t)arow * K;

    v8f acc[4];
#pragma unroll
    for (int ct = 0; ct < 4; ++ct) acc[ct] = (v8f){0.f,0.f,0.f,0.f,0.f,0.f,0.f,0.f};

    const int bk = half * 8;          // lane half h carries K = {8h..8h+7, 16+8h..16+8h+7} of the B tile

    // A-fragment pipeline prologue (16-bit A-matrix 16x32 layout:
    // lane half 0: K = {0..7, 16..23}; half 1: K = {8..15, 24..31})
    v8bf alo = *(const v8bf*)(Arow + half * 8);
    v8bf ahi = *(const v8bf*)(Arow + 16 + half * 8);

    for (int kp = 0; kp < K; kp += KP) {
        __builtin_prefetch(Arow + kp + KP, 0, 3);   // -> global_prefetch_b8 (near)
        __syncthreads();                            // panel reuse fence
        // cooperative B-panel fill: 64 rows x KP cols, 16B per thread per step
#pragma unroll
        for (int t = threadIdx.x; t < (64 * KP) / 8; t += 256) {
            const int idx  = t * 8;
            const int brow = idx / KP;
            const int kk   = idx % KP;
            const __bf16* gp = W + (size_t)(col0 + brow) * K + kp + kk;
            __bf16*       lp = sB + brow * SP + kk;
#if USE_ASYNC_LDS
            __builtin_amdgcn_global_load_async_to_lds_b128(
                (__attribute__((address_space(1))) v4i*)gp,
                (__attribute__((address_space(3))) v4i*)lp, 0, 0);
#else
            *(v8bf*)lp = *(const v8bf*)gp;
#endif
        }
#if USE_ASYNC_LDS
#if __has_builtin(__builtin_amdgcn_s_wait_asynccnt)
        __builtin_amdgcn_s_wait_asynccnt(0);
#else
        asm volatile("s_wait_asynccnt 0x0" ::: "memory");
#endif
#endif
        __syncthreads();

        for (int k0 = 0; k0 < KP; k0 += 32) {
            // issue next A-fragment loads before consuming current ones
            int knext = kp + k0 + 32;
            if (knext >= K) knext = K - 32;         // tail: redundant reload, harmless
            v8bf nlo = *(const v8bf*)(Arow + knext + half * 8);
            v8bf nhi = *(const v8bf*)(Arow + knext + 16 + half * 8);

            v16bf a;
#pragma unroll
            for (int i = 0; i < 8; ++i) { a[i] = alo[i]; a[8 + i] = ahi[i]; }

#pragma unroll
            for (int ct = 0; ct < 4; ++ct) {
                // B fragment from LDS: lane holds column (col0+ct*16+r),
                // K values k0+bk..+7 and k0+16+bk..+7 (matches A layout)
                const __bf16* bp = sB + (ct * 16 + r) * SP + k0 + bk;
                v8bf blo = *(const v8bf*)bp;
                v8bf bhi = *(const v8bf*)(bp + 16);
                v16bf b;
#pragma unroll
                for (int i = 0; i < 8; ++i) { b[i] = blo[i]; b[8 + i] = bhi[i]; }
                acc[ct] = __builtin_amdgcn_wmma_f32_16x16x32_bf16(
                    false, a, false, b, (short)0, acc[ct], false, false);
            }
            alo = nlo; ahi = nhi;
        }
    }

#pragma unroll
    for (int ct = 0; ct < 4; ++ct) {
        const int ccol = col0 + ct * 16 + r;
        const float bv = bias ? bias[ccol] : 0.f;
#pragma unroll
        for (int v = 0; v < 8; ++v) {
            const int rr = row0 + half * 8 + v;   // C/D layout: lanes>=16 hold M=8..15
            if (rr < M) C[(size_t)rr * Nc + ccol] = acc[ct][v] + bv;
        }
    }
}

// ---------------------------------------------------------------------------
// attention pass A: per (edge, head) logit + segment-max via encoded atomicMax
// ---------------------------------------------------------------------------
__device__ __forceinline__ unsigned f2mono(float f) {
    unsigned u = __float_as_uint(f);
    return (u & 0x80000000u) ? ~u : (u | 0x80000000u);
}
__device__ __forceinline__ float mono2f(unsigned u) {
    return __uint_as_float((u & 0x80000000u) ? (u & 0x7fffffffu) : ~u);
}

__global__ void edge_logits_k(const int* __restrict__ ei, const float* __restrict__ ea,
                              const float* __restrict__ xl, const float* __restrict__ xr,
                              const float* __restrict__ We, const float* __restrict__ att,
                              float* __restrict__ logits, unsigned* __restrict__ mbuf,
                              int H, int Cc) {
    long long t = (long long)blockIdx.x * blockDim.x + threadIdx.x;
    if (t >= (long long)GE * H) return;
    const int e = (int)(t / H), h = (int)(t % H);
    const int src = ei[e], dst = ei[GE + e];
    const float eav = ea[e];
    const int hc = H * Cc;
    const float* xls  = xl  + (size_t)src * hc + h * Cc;
    const float* xrd  = xr  + (size_t)dst * hc + h * Cc;
    const float* Weh  = We  + h * Cc;
    const float* atth = att + h * Cc;
    float acc = 0.f;
    for (int c = 0; c < Cc; ++c) {
        float z = xls[c] + xrd[c] + eav * Weh[c];
        z = z > 0.f ? z : 0.2f * z;               // leaky_relu 0.2
        acc += z * atth[c];
    }
    logits[t] = acc;
    atomicMax(&mbuf[(size_t)dst * H + h], f2mono(acc));
}

// ---------------------------------------------------------------------------
// attention pass B: ex = exp(logit - max[dst]); denom[dst] += ex (in place)
// ---------------------------------------------------------------------------
__global__ void edge_exp_k(const int* __restrict__ ei, float* __restrict__ logits,
                           const unsigned* __restrict__ mbuf, float* __restrict__ denom,
                           int H) {
    long long t = (long long)blockIdx.x * blockDim.x + threadIdx.x;
    if (t >= (long long)GE * H) return;
    const int e = (int)(t / H), h = (int)(t % H);
    const int dst = ei[GE + e];
    const float m = mono2f(mbuf[(size_t)dst * H + h]);
    const float ex = __expf(logits[t] - m);
    logits[t] = ex;
    atomicAdd(&denom[(size_t)dst * H + h], ex);
}

// ---------------------------------------------------------------------------
// attention pass C: agg[dst,j] += xl[src,j] * alpha(e, head(j))
// ---------------------------------------------------------------------------
__global__ void edge_agg_k(const int* __restrict__ ei, const float* __restrict__ exw,
                           const float* __restrict__ denom, const float* __restrict__ xl,
                           float* __restrict__ agg, int H, int Cc) {
    const int hc = H * Cc;
    long long t = (long long)blockIdx.x * blockDim.x + threadIdx.x;
    if (t >= (long long)GE * hc) return;
    const int e = (int)(t / hc), j = (int)(t % hc), h = j / Cc;
    const int src = ei[e], dst = ei[GE + e];
    const float alpha = exw[(size_t)e * H + h] / denom[(size_t)dst * H + h];
    atomicAdd(&agg[(size_t)dst * hc + j], xl[(size_t)src * hc + j] * alpha);
}

// ---------------------------------------------------------------------------
// per-node: (+bias) -> LayerNorm -> leaky_relu(0.01) -> + residual
// one wave32 per node; nd in {64, 256}
// ---------------------------------------------------------------------------
__global__ __launch_bounds__(256)
void node_finalize_k(const float* __restrict__ agg, const float* __restrict__ bias,
                     const float* __restrict__ g, const float* __restrict__ beta,
                     const float* __restrict__ res, float* __restrict__ out, int nd) {
    const int wave = threadIdx.x >> 5, lane = threadIdx.x & 31;
    const int n = blockIdx.x * 8 + wave;
    if (n >= GN) return;
    const int cnt = nd >> 5;                        // elems per lane (2 or 8)
    const float* a = agg + (size_t)n * nd;
    float v[8];
    float s = 0.f;
    for (int i = 0; i < cnt; ++i) {
        const int j = lane + (i << 5);
        v[i] = a[j] + bias[j];
        s += v[i];
    }
#pragma unroll
    for (int off = 16; off > 0; off >>= 1) s += __shfl_xor(s, off, 32);
    const float mu = s / nd;
    float vs = 0.f;
    for (int i = 0; i < cnt; ++i) { const float d = v[i] - mu; vs += d * d; }
#pragma unroll
    for (int off = 16; off > 0; off >>= 1) vs += __shfl_xor(vs, off, 32);
    const float inv = rsqrtf(vs / nd + 1e-5f);
    for (int i = 0; i < cnt; ++i) {
        const int j = lane + (i << 5);
        float y = (v[i] - mu) * inv * g[j] + beta[j];
        y = y > 0.f ? y : 0.01f * y;
        out[(size_t)n * nd + j] = y + res[(size_t)n * nd + j];
    }
}

// ---------------------------------------------------------------------------
// host orchestration
// ---------------------------------------------------------------------------
static inline size_t align256(size_t x) { return (x + 255) & ~(size_t)255; }

extern "C" void kernel_launch(void* const* d_in, const int* in_sizes, int n_in,
                              void* d_out, int out_size, void* d_ws, size_t ws_size,
                              hipStream_t stream) {
    (void)in_sizes; (void)n_in; (void)out_size; (void)ws_size;
    const int N = GN, E = GE;

    const int*   ei = (const int*)d_in[0];
    const float* ea = (const float*)d_in[1];
    const float* x  = (const float*)d_in[2];
    // layer param accessor: 9 per layer starting at index 3
    auto P = [&](int layer, int idx) { return (const float*)d_in[3 + layer * 9 + idx]; };
    const float* Rw0 = (const float*)d_in[30];
    const float* Rb0 = (const float*)d_in[31];
    const float* Rw2 = (const float*)d_in[32];
    const float* Rb2 = (const float*)d_in[33];

    // workspace layout
    char* base = (char*)d_ws;
    size_t off = 0;
    auto alloc = [&](size_t bytes) -> void* {
        void* p = base + off; off = align256(off + bytes); return p;
    };
    __bf16*   xb    = (__bf16*)alloc((size_t)N * 768 * 2);
    float*    xl    = (float*)alloc((size_t)N * 256 * 4);
    float*    xr    = (float*)alloc((size_t)N * 256 * 4);
    float*    agg   = (float*)alloc((size_t)N * 256 * 4);
    float*    h0    = (float*)alloc((size_t)N * 256 * 4);
    float*    h1    = (float*)alloc((size_t)N * 256 * 4);
    float*    res   = (float*)alloc((size_t)N * 256 * 4);
    float*    lg    = (float*)alloc((size_t)E * 4 * 4);
    unsigned* mbuf  = (unsigned*)alloc((size_t)N * 4 * 4);
    float*    denom = (float*)alloc((size_t)N * 4 * 4);
    __bf16* Wl0b = (__bf16*)alloc((size_t)256 * 768 * 2);
    __bf16* Wr0b = (__bf16*)alloc((size_t)256 * 768 * 2);
    __bf16* Rw0b = (__bf16*)alloc((size_t)256 * 768 * 2);
    __bf16* Wl1b = (__bf16*)alloc((size_t)256 * 256 * 2);
    __bf16* Wr1b = (__bf16*)alloc((size_t)256 * 256 * 2);
    __bf16* Wl2b = (__bf16*)alloc((size_t)64 * 256 * 2);
    __bf16* Wr2b = (__bf16*)alloc((size_t)64 * 256 * 2);
    __bf16* Rw2b = (__bf16*)alloc((size_t)64 * 256 * 2);

    auto cvt = [&](const float* s, __bf16* d, long long n) {
        f32_to_bf16_k<<<dim3((unsigned)((n + 255) / 256)), 256, 0, stream>>>(s, d, n);
    };
    auto fill = [&](void* p, unsigned v, long long n) {
        fill_u32_k<<<dim3((unsigned)((n + 255) / 256)), 256, 0, stream>>>((unsigned*)p, v, n);
    };
    auto gemm = [&](const __bf16* A, const __bf16* W, const float* b, float* C,
                    int M, int K, int Nc) {
        dim3 grid((M + 127) / 128, Nc / 64);
        gemm_bf16_wmma_k<<<grid, 256, 0, stream>>>(A, W, b, C, M, K, Nc);
    };

    const unsigned NEG_INF_MONO = 0x007FFFFFu;   // f2mono(-inf)

    // ---- weight conversions (small, once) ----
    cvt(P(0,0), Wl0b, 256*768); cvt(P(0,2), Wr0b, 256*768); cvt(Rw0, Rw0b, 256*768);
    cvt(P(1,0), Wl1b, 256*256); cvt(P(1,2), Wr1b, 256*256);
    cvt(P(2,0), Wl2b, 64*256);  cvt(P(2,2), Wr2b, 64*256);  cvt(Rw2, Rw2b, 64*256);

    const long long EH4  = (long long)E * 4;
    const long long EHC  = (long long)E * 256;
    const long long EH1  = (long long)E;
    const long long EHC2 = (long long)E * 64;

    // ================= layer 0: cin=768, hc=256, H=4 =================
    cvt(x, xb, (long long)N * 768);
    gemm(xb, Wl0b, P(0,1), xl, N, 768, 256);
    gemm(xb, Wr0b, P(0,3), xr, N, 768, 256);
    gemm(xb, Rw0b, Rb0,    res, N, 768, 256);
    fill(mbuf, NEG_INF_MONO, (long long)N * 4);
    fill(denom, 0u, (long long)N * 4);
    fill(agg, 0u, (long long)N * 256);
    edge_logits_k<<<dim3((unsigned)((EH4 + 255)/256)), 256, 0, stream>>>(
        ei, ea, xl, xr, P(0,4), P(0,5), lg, mbuf, 4, 64);
    edge_exp_k<<<dim3((unsigned)((EH4 + 255)/256)), 256, 0, stream>>>(ei, lg, mbuf, denom, 4);
    edge_agg_k<<<dim3((unsigned)((EHC + 255)/256)), 256, 0, stream>>>(ei, lg, denom, xl, agg, 4, 64);
    node_finalize_k<<<dim3((N + 7)/8), 256, 0, stream>>>(agg, P(0,6), P(0,7), P(0,8), res, h0, 256);

    // ================= layer 1: cin=256, hc=256, H=4 =================
    cvt(h0, xb, (long long)N * 256);
    gemm(xb, Wl1b, P(1,1), xl, N, 256, 256);
    gemm(xb, Wr1b, P(1,3), xr, N, 256, 256);
    fill(mbuf, NEG_INF_MONO, (long long)N * 4);
    fill(denom, 0u, (long long)N * 4);
    fill(agg, 0u, (long long)N * 256);
    edge_logits_k<<<dim3((unsigned)((EH4 + 255)/256)), 256, 0, stream>>>(
        ei, ea, xl, xr, P(1,4), P(1,5), lg, mbuf, 4, 64);
    edge_exp_k<<<dim3((unsigned)((EH4 + 255)/256)), 256, 0, stream>>>(ei, lg, mbuf, denom, 4);
    edge_agg_k<<<dim3((unsigned)((EHC + 255)/256)), 256, 0, stream>>>(ei, lg, denom, xl, agg, 4, 64);
    node_finalize_k<<<dim3((N + 7)/8), 256, 0, stream>>>(agg, P(1,6), P(1,7), P(1,8), h0, h1, 256);

    // ================= layer 2: cin=256, hc=64, H=1 ==================
    cvt(h1, xb, (long long)N * 256);
    gemm(xb, Wl2b, P(2,1), xl, N, 256, 64);
    gemm(xb, Wr2b, P(2,3), xr, N, 256, 64);
    gemm(xb, Rw2b, Rb2,    res, N, 256, 64);
    fill(mbuf, NEG_INF_MONO, (long long)N);
    fill(denom, 0u, (long long)N);
    fill(agg, 0u, (long long)N * 64);
    edge_logits_k<<<dim3((unsigned)((EH1 + 255)/256)), 256, 0, stream>>>(
        ei, ea, xl, xr, P(2,4), P(2,5), lg, mbuf, 1, 64);
    edge_exp_k<<<dim3((unsigned)((EH1 + 255)/256)), 256, 0, stream>>>(ei, lg, mbuf, denom, 1);
    edge_agg_k<<<dim3((unsigned)((EHC2 + 255)/256)), 256, 0, stream>>>(ei, lg, denom, xl, agg, 1, 64);
    node_finalize_k<<<dim3((N + 7)/8), 256, 0, stream>>>(agg, P(2,6), P(2,7), P(2,8), res,
                                                         (float*)d_out, 64);
}
